// BinaryMechanismSSM_12463995093229
// MI455X (gfx1250) — matrix operand order repaired and checked by reading, not compiled, not run on hardware
//
#include <hip/hip_runtime.h>

// ---------------------------------------------------------------------------
// BinaryMechanismSSM on MI455X (gfx1250, wave32, WMMA)
// 8 independent WGs (16 batch rows each); full f16 state in LDS; fused input
// projections; f16 WMMA w/ fp32 accumulate. Weights repacked once into
// fragment-contiguous f16 layout so every B-fragment load is a coalesced
// contiguous 1KB wave read. GEMM passes serialized through LDS staging
// buffers to keep VGPR pressure low (no spills).
// ---------------------------------------------------------------------------

typedef __attribute__((ext_vector_type(16))) _Float16 v16h;
typedef __attribute__((ext_vector_type(8)))  _Float16 v8h;
typedef __attribute__((ext_vector_type(8)))  float    v8f;

union F16Frag { v16h v; v8h h[2]; };

constexpr int kT = 1024;
constexpr int kI = 256;
constexpr int kS = 512;
constexpr int kRows = 16;          // batch rows per workgroup
constexpr int X_LD  = kI + 8;      // padded LDS rows (halves): bank-staggered
constexpr int S_LD  = kS + 8;
constexpr int F_LD  = kS + 16;     // padded f32 LDS rows (floats)

// f16 weight workspace layout (offsets in halves); each region holds the
// weight repacked as fragment-contiguous blocks of 512 halves (16 cols x 32 K).
constexpr int OFF_A0 = 0;
constexpr int OFF_A1 = OFF_A0 + kS * kS;
constexpr int OFF_B0 = OFF_A1 + kS * kS;
constexpr int OFF_B1 = OFF_B0 + kS * kI;
constexpr int OFF_G  = OFF_B1 + kS * kI;
constexpr int TOT_W  = OFF_G  + kS * kI;   // 917504 halves = 1.75 MB

// dynamic LDS partition (bytes)
constexpr int SM_STATE = kRows * S_LD * 2;                   // 16640
constexpr int SM_X     = kRows * X_LD * 2;                   // 8448
constexpr int SM_BXROW = kRows * F_LD * 4;                   // 33792 per f32 buffer
constexpr size_t SMEM_BYTES = SM_STATE + SM_X + 3 * SM_BXROW + 3 * kS * 4;

// A fragment (16x32 f16) from LDS: lane holds row m=lane%16.
// lanes 0-15: K = kk+{0..15}; lanes 16-31: K = kk+{16..31}
__device__ inline v16h load_frag_a(const _Float16* base, int ld, int m, int kk, int selA) {
  const _Float16* p = base + m * ld + kk + 2 * selA;
  F16Frag f;
  f.h[0] = *(const v8h*)(p);
  f.h[1] = *(const v8h*)(p + 8);
  return f.v;
}

// B fragment (32x16 f16) from the repacked weight stream: fragment
// (tileIdx, kk/32) is a contiguous 1KB block; lane L reads halves
// [L*16 .. L*16+15] (lane = hl*16+nl uses the same k order as load_frag_a:
// lane holds column n=L%16, K = kk + (L<16 ? 0..15 : 16..31)).
__device__ inline v16h load_frag_b(const _Float16* __restrict__ Wf, int K,
                                   int tileIdx, int kk, int lane) {
  const _Float16* p = Wf + ((tileIdx * (K >> 5) + (kk >> 5)) << 9) + (lane << 4);
  F16Frag f;
  f.h[0] = *(const v8h*)(p);
  f.h[1] = *(const v8h*)(p + 8);
  return f.v;
}

// One GEMM pass: acc[0..1] += sA(16xK) @ W^T for this wave's 2 column tiles.
__device__ inline void gemm_pass1(const _Float16* __restrict__ Wf, int K,
                                  const _Float16* sA, int lda,
                                  int tile0, int lane, int nl, int selA,
                                  v8f acc[2]) {
#pragma unroll 2
  for (int kk = 0; kk < K; kk += 32) {
    v16h a = load_frag_a(sA, lda, nl, kk, selA);
#pragma unroll
    for (int c = 0; c < 2; ++c) {
      acc[c] = __builtin_amdgcn_wmma_f32_16x16x32_f16(
          false, a, false, load_frag_b(Wf, K, tile0 + c, kk, lane),
          (short)0, acc[c], false, false);
    }
  }
}

// Dual-matrix pass sharing the A fragment (recurrent GEMMs vs A0,A1).
__device__ inline void gemm_pass2(const _Float16* __restrict__ W0f,
                                  const _Float16* __restrict__ W1f, int K,
                                  const _Float16* sA, int lda,
                                  int tile0, int lane, int nl, int selA,
                                  v8f acc0[2], v8f acc1[2]) {
#pragma unroll 2
  for (int kk = 0; kk < K; kk += 32) {
    v16h a = load_frag_a(sA, lda, nl, kk, selA);
#pragma unroll
    for (int c = 0; c < 2; ++c) {
      acc0[c] = __builtin_amdgcn_wmma_f32_16x16x32_f16(
          false, a, false, load_frag_b(W0f, K, tile0 + c, kk, lane),
          (short)0, acc0[c], false, false);
      acc1[c] = __builtin_amdgcn_wmma_f32_16x16x32_f16(
          false, a, false, load_frag_b(W1f, K, tile0 + c, kk, lane),
          (short)0, acc1[c], false, false);
    }
  }
}

// Repack W[n][k] (f32, row-major) -> f16 fragment-contiguous blocks:
// dest = ((n/16)*(K/32) + k/32)*512 + ((k%32)/16)*256 + (n%16)*16 + (k%16)
__device__ inline int frag_index(int n, int k, int K) {
  return (((n >> 4) * (K >> 5) + (k >> 5)) << 9) + (((k >> 4) & 1) << 8) +
         ((n & 15) << 4) + (k & 15);
}

__global__ void cvt_weights(const float* __restrict__ A0, const float* __restrict__ A1,
                            const float* __restrict__ B0, const float* __restrict__ B1,
                            const float* __restrict__ G,  _Float16* __restrict__ w) {
  int i = blockIdx.x * blockDim.x + threadIdx.x;
  if (i < OFF_B0) {            // A0, A1: [512][512]
    const float* src = (i < OFF_A1) ? A0 : A1;
    int li = (i < OFF_A1) ? i : i - OFF_A1;
    int base = (i < OFF_A1) ? OFF_A0 : OFF_A1;
    int n = li / kS, k = li % kS;
    w[base + frag_index(n, k, kS)] = (_Float16)src[li];
  } else if (i < TOT_W) {      // B0, B1, gate: [512][256]
    const float* src; int li, base;
    if (i < OFF_B1)      { src = B0; li = i - OFF_B0; base = OFF_B0; }
    else if (i < OFF_G)  { src = B1; li = i - OFF_B1; base = OFF_B1; }
    else                 { src = G;  li = i - OFF_G;  base = OFF_G;  }
    int n = li / kI, k = li % kI;
    w[base + frag_index(n, k, kI)] = (_Float16)src[li];
  }
}

__global__ __launch_bounds__(512) void ssm_scan(
    const float* __restrict__ x,     const float* __restrict__ s0,
    const _Float16* __restrict__ wsh,
    const float* __restrict__ b0b,   const float* __restrict__ b1b,
    const float* __restrict__ gbias, const float* __restrict__ alphap,
    const int* __restrict__ zp,      float* __restrict__ out) {
  extern __shared__ __align__(16) char smem[];
  _Float16* s_state = (_Float16*)smem;                    // [16][S_LD] f16
  _Float16* s_x     = s_state + kRows * S_LD;             // [16][X_LD] f16
  float*    s_bx0   = (float*)(s_x + kRows * X_LD);       // [16][F_LD] f32
  float*    s_bx1   = s_bx0 + kRows * F_LD;
  float*    s_g     = s_bx1 + kRows * F_LD;
  float*    s_b0    = s_g + kRows * F_LD;                 // biases [512] each
  float*    s_b1    = s_b0 + kS;
  float*    s_bg    = s_b1 + kS;

  const int tid  = threadIdx.x;        // 512 threads = 16 wave32 waves
  const int lane = tid & 31;
  const int wid  = tid >> 5;           // wave owns 32 output columns
  const int hl   = (lane >> 4) & 1;
  const int nl   = lane & 15;
  const int selA = hl ? 8 : 0;
  const int tile0 = wid * 2;           // first of this wave's 2 column tiles
  const int cb    = wid * 32;          // wave column base
  const int brow  = blockIdx.x * kRows;

  const _Float16* A0h = wsh + OFF_A0;
  const _Float16* A1h = wsh + OFF_A1;
  const _Float16* B0h = wsh + OFF_B0;
  const _Float16* B1h = wsh + OFF_B1;
  const _Float16* Gh  = wsh + OFF_G;

  const float alpha = alphap[0];
  const int   z     = zp[0];

  // biases -> LDS (blockDim == kS)
  s_b0[tid] = b0b[tid];
  s_b1[tid] = b1b[tid];
  s_bg[tid] = gbias[tid];

  // s0 -> LDS f16 state, and out[:,0,:]
  for (int i = tid; i < kRows * kS; i += 512) {
    int r = i >> 9, c = i & (kS - 1);
    float v = s0[(size_t)(brow + r) * kS + c];
    s_state[r * S_LD + c] = (_Float16)v;
    out[(size_t)(brow + r) * (kT + 1) * kS + c] = v;
  }
  __syncthreads();

  const v8f vzero = {0.f, 0.f, 0.f, 0.f, 0.f, 0.f, 0.f, 0.f};

  for (int t = 0; t < kT; ++t) {
    // ---- stage x(t) tile [16 x 256] fp32 -> f16 LDS (coalesced float4) ----
    for (int i = tid; i < kRows * (kI / 4); i += 512) {
      int r = i >> 6, c4 = i & 63;
      const float4 xv = ((const float4*)(x + ((size_t)(brow + r) * kT + t) * kI))[c4];
      int c = c4 * 4;
      s_x[r * X_LD + c + 0] = (_Float16)xv.x;
      s_x[r * X_LD + c + 1] = (_Float16)xv.y;
      s_x[r * X_LD + c + 2] = (_Float16)xv.z;
      s_x[r * X_LD + c + 3] = (_Float16)xv.w;
    }
    // prefetch next timestep's x tile (global_prefetch_b8)
    if (t + 1 < kT)
      __builtin_prefetch(x + ((size_t)(brow + wid) * kT + (t + 1)) * kI + lane * 8, 0, 1);
    __syncthreads();

    // ---- input projections, one matrix at a time (low VGPR pressure) ----
    {
      v8f acc[2] = {vzero, vzero};
      gemm_pass1(B0h, kI, s_x, X_LD, tile0, lane, nl, selA, acc);
#pragma unroll
      for (int c = 0; c < 2; ++c) {
        int col = cb + c * 16 + nl;
        float bb = s_b0[col];
#pragma unroll
        for (int r = 0; r < 8; ++r) s_bx0[(r + 8 * hl) * F_LD + col] = acc[c][r] + bb;
      }
    }
    {
      v8f acc[2] = {vzero, vzero};
      gemm_pass1(B1h, kI, s_x, X_LD, tile0, lane, nl, selA, acc);
#pragma unroll
      for (int c = 0; c < 2; ++c) {
        int col = cb + c * 16 + nl;
        float bb = s_b1[col];
#pragma unroll
        for (int r = 0; r < 8; ++r) s_bx1[(r + 8 * hl) * F_LD + col] = acc[c][r] + bb;
      }
    }
    {
      v8f acc[2] = {vzero, vzero};
      gemm_pass1(Gh, kI, s_x, X_LD, tile0, lane, nl, selA, acc);
#pragma unroll
      for (int c = 0; c < 2; ++c) {
        int col = cb + c * 16 + nl;
        float bb = s_bg[col];
#pragma unroll
        for (int r = 0; r < 8; ++r)
          s_g[(r + 8 * hl) * F_LD + col] = 1.0f / (1.0f + __expf(-(acc[c][r] + bb)));
      }
    }

    // ---- recurrent GEMMs: s @ A0^T and s @ A1^T (shared A fragment) ----
    v8f aF0[2] = {vzero, vzero}, aF1[2] = {vzero, vzero};
    gemm_pass2(A0h, A1h, kS, s_state, S_LD, tile0, lane, nl, selA, aF0, aF1);
    __syncthreads();   // all waves done reading s_state

    // ---- elementwise gate + state update + global store ----
#pragma unroll
    for (int c = 0; c < 2; ++c) {
      int col = cb + c * 16 + nl;
#pragma unroll
      for (int r = 0; r < 8; ++r) {
        int m = r + 8 * hl;   // D-matrix row for VGPR r (lanes 16-31 -> +8)
        float f0 = tanhf(aF0[c][r] + s_bx0[m * F_LD + col]);
        float f1 = tanhf(aF1[c][r] + s_bx1[m * F_LD + col]);
        float f  = z ? ((1.0f - alpha) * f0 + alpha * f1) : f0;
        float g  = s_g[m * F_LD + col];
        float so = (float)s_state[m * S_LD + col];
        float sn = fmaf(g, f - so, so);            // g*f + (1-g)*so
        s_state[m * S_LD + col] = (_Float16)sn;
        out[((size_t)(brow + m) * (kT + 1) + (t + 1)) * kS + col] = sn;
      }
    }
    __syncthreads();   // state writes visible before next step
  }
}

extern "C" void kernel_launch(void* const* d_in, const int* in_sizes, int n_in,
                              void* d_out, int out_size, void* d_ws, size_t ws_size,
                              hipStream_t stream) {
  const float* x      = (const float*)d_in[0];
  const float* s0     = (const float*)d_in[1];
  const float* A0w    = (const float*)d_in[2];
  const float* B0w    = (const float*)d_in[3];
  const float* B0bias = (const float*)d_in[4];
  const float* A1w    = (const float*)d_in[5];
  const float* B1w    = (const float*)d_in[6];
  const float* B1bias = (const float*)d_in[7];
  const float* Gw     = (const float*)d_in[8];
  const float* Gbias  = (const float*)d_in[9];
  const float* alphap = (const float*)d_in[10];
  const int*   zp     = (const int*)d_in[11];
  float*       out    = (float*)d_out;
  _Float16*    wsh    = (_Float16*)d_ws;

  cvt_weights<<<(TOT_W + 511) / 512, 512, 0, stream>>>(A0w, A1w, B0w, B1w, Gw, wsh);
  ssm_scan<<<dim3(128 / kRows), dim3(512), SMEM_BYTES, stream>>>(
      x, s0, wsh, B0bias, B1bias, Gbias, alphap, zp, out);
}
